// VectorQuantizer_36867999269233
// MI455X (gfx1250) — matrix operand order repaired and checked by reading, not compiled, not run on hardware
//
#include <hip/hip_runtime.h>
#include <hip/hip_bf16.h>
#include <float.h>
#include <math.h>
#include <stdint.h>

// ---------------------------------------------------------------------------
// VQ (VectorQuantizer): z [16,256,32,32] f32, codebook [8192,256] f32
//   N = 16*32*32 = 16384 rows, C = 256 channels, K = 8192 codes
// Outputs (flat, float): quantized [4194304] | indices [16384] | perplexity,
//   loss_vq, loss_commit (3 scalars)
// argmin_k ||z-e_k||^2 == argmin_k ( 0.5*||e_k||^2 - z.e_k )  (z-norm const)
// Cross-term via v_wmma_f32_16x16x32_bf16 (4 independent accumulator chains;
// DS-read/WMMA interleave pinned with sched_group_barrier); codebook chunks
// double-buffered into LDS with global_load_async_to_lds_b128 (ASYNCcnt).
// ---------------------------------------------------------------------------

typedef __attribute__((ext_vector_type(16))) __bf16 v16bf;
typedef __attribute__((ext_vector_type(8)))  __bf16 v8bf;
typedef __attribute__((ext_vector_type(8)))  float  v8f;

#define VQ_C     256
#define VQ_K     8192
#define VQ_N     16384
#define VQ_HW    1024          // 32*32
#define VQ_CHW   262144        // 256*1024
#define VQ_NUMEL 4194304       // 16*256*32*32
#define ROWP     264           // padded LDS row stride (elements): 528B = 132dw

#define SGB __builtin_amdgcn_sched_group_barrier
#define SGB_DSREAD 0x100
#define SGB_WMMA   0x008

// ---- prep: z [B,C,H,W] f32 -> z_bf16 [N,C] (NHWC-flat) ---------------------
__global__ void vq_prep_z(const float* __restrict__ z,
                          __bf16* __restrict__ zb) {
  const int n  = blockIdx.x;          // row in flat_z
  const int ch = threadIdx.x;         // channel
  const int b = n >> 10, rem = n & 1023;
  float v = z[(size_t)b * VQ_CHW + (size_t)ch * VQ_HW + rem];
  zb[(size_t)n * VQ_C + ch] = (__bf16)v;
}

// ---- prep: codebook [K,C] f32 -> bf16 copy + 0.5*||e_k||^2 -----------------
__global__ void vq_prep_cb(const float* __restrict__ cbf,
                           __bf16* __restrict__ cbb,
                           float* __restrict__ cbn_h) {
  const int k  = blockIdx.x;
  const int ch = threadIdx.x;
  float v = cbf[(size_t)k * VQ_C + ch];
  cbb[(size_t)k * VQ_C + ch] = (__bf16)v;
  __shared__ float red[256];
  red[ch] = v * v;
  __syncthreads();
  for (int s = 128; s > 0; s >>= 1) {
    if (ch < s) red[ch] += red[ch + s];
    __syncthreads();
  }
  if (ch == 0) cbn_h[k] = 0.5f * red[0];
}

// ---- zero counts + loss accumulator ---------------------------------------
__global__ void vq_zero(float* __restrict__ counts, float* __restrict__ lacc) {
  int i = blockIdx.x * blockDim.x + threadIdx.x;
  if (i < VQ_K) counts[i] = 0.0f;
  if (i == VQ_K) *lacc = 0.0f;
}

// ---- main: WMMA distance search + argmin + gather/losses -------------------
// Block = 256 threads = 8 waves; wave w owns 16 z-rows; block owns 128 rows.
// Codebook swept in 32-code chunks, double-buffered in LDS via async DMA.
__global__ __launch_bounds__(256, 1)
void vq_argmin(const __bf16* __restrict__ zb, const __bf16* __restrict__ cb,
               const float* __restrict__ cbn_h,
               const float* __restrict__ z, const float* __restrict__ codebook,
               float* __restrict__ out_q, float* __restrict__ out_idx,
               float* __restrict__ counts, float* __restrict__ lacc) {
  __shared__ __align__(16) __bf16 sB[2][32 * ROWP];  // 2 x 16,896 B
  __shared__ float s_cbn[VQ_K];                      // 32 KB half-norms
  __shared__ int   s_idx[128];
  __shared__ float s_red[256];

  const int tid  = threadIdx.x;
  const int lane = tid & 31;
  const int wv   = tid >> 5;
  const int nBlk = blockIdx.x * 128;       // block row base
  const int n0   = nBlk + wv * 16;         // wave row base (16 rows)
  const int nlo  = lane & 15;
  const bool hi  = lane >= 16;

  // Stage all half-norms into LDS once (coalesced, conflict-free reads later).
  for (int i = tid; i < VQ_K; i += 256) s_cbn[i] = cbn_h[i];

  // --- Preload A (z tile 16x256) into 8 x v16bf, ISA VGPR-layout swizzled.
  // Lane: row m = lane%16; K set = {kb..kb+7, kb+16..kb+23}, kb = hi?8:0.
  v16bf A[8];
  {
    const int kb = hi ? 8 : 0;
    const __bf16* rowp = zb + (size_t)(n0 + nlo) * VQ_C;
#pragma unroll
    for (int cc = 0; cc < 8; ++cc) {
      union { v16bf v; v8bf h[2]; } u;
      u.h[0] = *(const v8bf*)(rowp + cc * 32 + kb);
      u.h[1] = *(const v8bf*)(rowp + cc * 32 + kb + 16);
      A[cc] = u.v;
    }
  }

  float best[8];
  int   bidx[8];
#pragma unroll
  for (int v = 0; v < 8; ++v) { best[v] = FLT_MAX; bidx[v] = 0; }

  // --- Async staging: 32 codes x 256ch bf16 = 16KB contiguous in global,
  // scattered into padded LDS rows (528B stride). 256 thr x 4 x 16B.
  const uint32_t ldsB0 = (uint32_t)(uintptr_t)&sB[0][0];
  const uint32_t ldsB1 = (uint32_t)(uintptr_t)&sB[1][0];
  auto stage = [&](int k0, uint32_t base) {
    const char* src = (const char*)(cb + (size_t)k0 * VQ_C);
#pragma unroll
    for (int it = 0; it < 4; ++it) {
      uint32_t off = (uint32_t)tid * 16u + (uint32_t)it * 4096u;  // chunk byte
      uint32_t dst = base + (off >> 9) * (ROWP * 2) + (off & 511u);
      const char* g = src + off;
      asm volatile("global_load_async_to_lds_b128 %0, %1, off"
                   :: "v"(dst), "v"(g) : "memory");
    }
  };

  stage(0, ldsB0);
  asm volatile("s_wait_asynccnt 0" ::: "memory");
  __syncthreads();   // covers s_cbn staging too

  // --- Sweep all K codes; 16 WMMA per 32-code chunk; DMA next chunk overlaps.
  union BU { v16bf v; v8bf h[2]; };
  for (int k0 = 0; k0 < VQ_K; k0 += 32) {
    const int cur = (k0 >> 5) & 1;
    const __bf16* sbuf = cur ? sB[1] : sB[0];
    const bool more = (k0 + 32) < VQ_K;
    if (more) stage(k0 + 32, cur ? ldsB0 : ldsB1);

    // Score bias terms (one ds_load_2addr_b32), hoisted ahead of WMMA block.
    const int   kn0 = k0 + nlo, kn1 = kn0 + 16;
    const float en0 = s_cbn[kn0], en1 = s_cbn[kn1];

    // B (32x16, k=channel, n=code): lane holds code (s*16 + lane%16),
    // channels cc*32 + {kb..kb+7, kb+16..kb+23}, kb = hi?8:0, from padded LDS rows.
    const __bf16* bp0 = sbuf + nlo * ROWP + (hi ? 8 : 0);    // codes k0+0..15
    const __bf16* bp1 = bp0 + 16 * ROWP;                     // codes k0+16..31

    // 4 independent WMMA chains: {subtile 0,1} x {even,odd cc}; B feeds are
    // software-pipelined one chunk ahead in separate register buffers.
    v8f accA0 = {}, accB0 = {}, accA1 = {}, accB1 = {};
    BU cur0, cur1, nxt0, nxt1;
    cur0.h[0] = *(const v8bf*)(bp0);  cur0.h[1] = *(const v8bf*)(bp0 + 16);
    cur1.h[0] = *(const v8bf*)(bp1);  cur1.h[1] = *(const v8bf*)(bp1 + 16);
#pragma unroll
    for (int cc = 0; cc < 8; ++cc) {
      if (cc < 7) {
        nxt0.h[0] = *(const v8bf*)(bp0 + (cc + 1) * 32);
        nxt0.h[1] = *(const v8bf*)(bp0 + (cc + 1) * 32 + 16);
        nxt1.h[0] = *(const v8bf*)(bp1 + (cc + 1) * 32);
        nxt1.h[1] = *(const v8bf*)(bp1 + (cc + 1) * 32 + 16);
      }
      if (cc & 1) {
        accB0 = __builtin_amdgcn_wmma_f32_16x16x32_bf16(
            false, A[cc], false, cur0.v, (short)0, accB0, false, false);
        accB1 = __builtin_amdgcn_wmma_f32_16x16x32_bf16(
            false, A[cc], false, cur1.v, (short)0, accB1, false, false);
      } else {
        accA0 = __builtin_amdgcn_wmma_f32_16x16x32_bf16(
            false, A[cc], false, cur0.v, (short)0, accA0, false, false);
        accA1 = __builtin_amdgcn_wmma_f32_16x16x32_bf16(
            false, A[cc], false, cur1.v, (short)0, accA1, false, false);
      }
      if (cc < 7) { cur0 = nxt0; cur1 = nxt1; }
    }

    // Pin the pipeline: cbn read + 4 B-loads ahead, then [4 DS][2 WMMA] x 7,
    // tail 2 WMMA. 33 DS-read slots, 16 WMMA slots per chunk.
    SGB(SGB_DSREAD, 1, 0);       // s_cbn ds_load_2addr
    SGB(SGB_DSREAD, 4, 0);       // cc=0 tiles (both subtiles)
#pragma unroll
    for (int i = 0; i < 7; ++i) {
      SGB(SGB_DSREAD, 4, 0);     // cc=i+1 tiles in flight
      SGB(SGB_WMMA,   2, 0);     // cc=i WMMAs retire behind them
    }
    SGB(SGB_WMMA, 2, 0);         // cc=7 WMMAs

#pragma unroll
    for (int v = 0; v < 8; ++v) {
      const float d0 = en0 - (accA0[v] + accB0[v]);       // 0.5||e||^2 - z.e
      if (d0 < best[v]) { best[v] = d0; bidx[v] = kn0; }  // first occurrence
      const float d1 = en1 - (accA1[v] + accB1[v]);
      if (d1 < best[v]) { best[v] = d1; bidx[v] = kn1; }
    }

    if (more) {
      asm volatile("s_wait_asynccnt 0" ::: "memory");
      __syncthreads();
    }
  }

  // --- Cross-lane argmin per row: butterfly within each 16-lane half.
#pragma unroll
  for (int v = 0; v < 8; ++v) {
    float d = best[v];
    int   i = bidx[v];
    for (int off = 8; off > 0; off >>= 1) {
      float d2 = __shfl_xor(d, off, 16);
      int   i2 = __shfl_xor(i, off, 16);
      if (d2 < d || (d2 == d && i2 < i)) { d = d2; i = i2; }
    }
    if (nlo == 0) s_idx[wv * 16 + v + (hi ? 8 : 0)] = i;  // lanes 0 and 16
  }
  __syncthreads();

  // --- Epilogue: gather quantized rows to [B,C,H,W], accumulate loss.
  const int b    = nBlk >> 10;
  const int roff = nBlk & 1023;    // 128 rows stay within one image
  float lsum = 0.0f;
  for (int e = tid; e < 128 * 256; e += 256) {
    const int c = e >> 7;          // channel
    const int r = e & 127;         // row offset within block (coalesced writes)
    const int idx = s_idx[r];
    const float q = codebook[(size_t)idx * VQ_C + c];
    const size_t oa = (size_t)b * VQ_CHW + (size_t)c * VQ_HW + roff + r;
    out_q[oa] = q;                 // straight-through value == quantized value
    const float dv = z[oa] - q;
    lsum += dv * dv;               // loss_vq == loss_commit numerically
  }
  s_red[tid] = lsum;
  __syncthreads();
  for (int s = 128; s > 0; s >>= 1) {
    if (tid < s) s_red[tid] += s_red[tid + s];
    __syncthreads();
  }
  if (tid == 0) atomicAdd(lacc, s_red[0]);

  if (tid < 128) {
    const int idx = s_idx[tid];
    out_idx[nBlk + tid] = (float)idx;
    atomicAdd(&counts[idx], 1.0f);
  }
}

// ---- finalize: perplexity + loss scalars ----------------------------------
__global__ void vq_finalize(const float* __restrict__ counts,
                            const float* __restrict__ lacc,
                            float* __restrict__ outs) {
  __shared__ float red[256];
  const int tid = threadIdx.x;
  float s = 0.0f;
  for (int i = tid; i < VQ_K; i += 256) {
    const float p  = counts[i] * (1.0f / VQ_N);
    const float pc = fmaxf(p, 1e-10f);
    s += p * logf(pc);             // probs * log(clip(probs))
  }
  red[tid] = s;
  __syncthreads();
  for (int st = 128; st > 0; st >>= 1) {
    if (tid < st) red[tid] += red[tid + st];
    __syncthreads();
  }
  if (tid == 0) {
    outs[0] = expf(-red[0]);                       // perplexity
    const float loss = *lacc * (1.0f / VQ_NUMEL);
    outs[1] = loss;                                // loss_vq
    outs[2] = loss;                                // loss_commit
  }
}

// ---------------------------------------------------------------------------
extern "C" void kernel_launch(void* const* d_in, const int* in_sizes, int n_in,
                              void* d_out, int out_size, void* d_ws, size_t ws_size,
                              hipStream_t stream) {
  const float* z        = (const float*)d_in[0];   // [16,256,32,32]
  const float* codebook = (const float*)d_in[1];   // [8192,256]
  float* out = (float*)d_out;

  // Workspace carve-up (bytes), all offsets 512B-aligned.
  char* ws = (char*)d_ws;
  __bf16* zb    = (__bf16*)(ws);                             // 8,388,608 B
  __bf16* cbb   = (__bf16*)(ws + 8388608);                   // 4,194,304 B
  float*  cbn_h = (float*)(ws + 12582912);                   //    32,768 B
  float*  cnts  = (float*)(ws + 12615680);                   //    32,768 B
  float*  lacc  = (float*)(ws + 12648448);                   //         4 B

  float* out_q   = out;                   // [4194304]
  float* out_idx = out + VQ_NUMEL;        // [16384]
  float* out_scl = out + VQ_NUMEL + VQ_N; // [3]

  vq_prep_z <<<VQ_N, 256, 0, stream>>>(z, zb);
  vq_prep_cb<<<VQ_K, 256, 0, stream>>>(codebook, cbb, cbn_h);
  vq_zero   <<<(VQ_K / 256) + 1, 256, 0, stream>>>(cnts, lacc);
  vq_argmin <<<VQ_N / 128, 256, 0, stream>>>(zb, cbb, cbn_h, z, codebook,
                                             out_q, out_idx, cnts, lacc);
  vq_finalize<<<1, 256, 0, stream>>>(cnts, lacc, out_scl);
}
